// OrderedAttention_89842125898335
// MI455X (gfx1250) — matrix operand order repaired and checked by reading, not compiled, not run on hardware
//
#include <hip/hip_runtime.h>
#include <math.h>

#define NUU 3
#define BB 16
#define NN 1024
#define DD 256
#define TOK (BB * NN)          // 16384 tokens per util
#define TILES (TOK / 16)       // 1024 16-token tiles per GEMM instance

typedef __attribute__((ext_vector_type(16))) _Float16 v16h;
typedef __attribute__((ext_vector_type(8)))  _Float16 v8h;
typedef __attribute__((ext_vector_type(8)))  float    v8f;

// ---------------------------------------------------------------------------
// Core: a 128-thread block (4 waves) computes E[t0..t0+15][0..255].
// Wave w owns output channels e in [w*64, w*64+64): 4 WMMA N-tiles = 4 v8f
// accumulators (32 VGPRs) -> no spills, high occupancy.
// C layout (ISA 7.12.2): element (m,n) -> lane = n + (m>=8)*16, vgpr = m&7.
// ---------------------------------------------------------------------------
static __device__ __forceinline__ void gemm4(const _Float16* __restrict__ A,
                                             const _Float16* __restrict__ W,
                                             const float* __restrict__ bias,
                                             int row0, int w, v8f c[4]) {
    const int l = threadIdx.x & 31;
    const int n = l & 15;
    const int g = l >> 4;
    const _Float16* arow  = A + (size_t)(row0 + n) * DD + g * 16;
    const _Float16* wbase = W + (size_t)(w * 64 + n) * DD + g * 16;

    #pragma unroll
    for (int et = 0; et < 4; ++et)
        #pragma unroll
        for (int q = 0; q < 8; ++q) c[et][q] = 0.0f;

    #pragma unroll
    for (int kc = 0; kc < 8; ++kc) {
        const int k0 = kc * 32;
        if (kc + 1 < 8) __builtin_prefetch(arow + k0 + 32, 0, 1);  // global_prefetch_b8
        v8h alo = *(const v8h*)(arow + k0);       // K = k0 + g*16 + 0..7
        v8h ahi = *(const v8h*)(arow + k0 + 8);   // K = k0 + g*16 + 8..15
        v16h a;
        #pragma unroll
        for (int i = 0; i < 8; ++i) { a[i] = alo[i]; a[i + 8] = ahi[i]; }
        #pragma unroll
        for (int et = 0; et < 4; ++et) {
            // B[k][e] = W[e][k]; lane covers e = w*64 + et*16 + n, K = k0+g*16+0..15
            v16h bf = *(const v16h*)(wbase + (size_t)et * 16 * DD + k0);
            c[et] = __builtin_amdgcn_wmma_f32_16x16x32_f16(
                false, a, false, bf, (short)0, c[et], false, false);
        }
    }
    #pragma unroll
    for (int et = 0; et < 4; ++et) {
        float bv = bias[w * 64 + et * 16 + n];
        #pragma unroll
        for (int q = 0; q < 8; ++q) c[et][q] += bv;
    }
}

// l2-normalize the 16 rows spread over the 4 waves. ssh[4][16] holds each
// wave's per-token partial sum of squares; tokens m = g*8 + r.
static __device__ __forceinline__ void l2norm_block(v8f c[4], int w, float ssh[4][16]) {
    const int l = threadIdx.x & 31;
    const int n = l & 15;
    const int g = l >> 4;
    #pragma unroll
    for (int r = 0; r < 8; ++r) {
        float ss = 0.0f;
        #pragma unroll
        for (int et = 0; et < 4; ++et) { float x = c[et][r]; ss += x * x; }
        ss += __shfl_xor(ss, 1, 32);
        ss += __shfl_xor(ss, 2, 32);
        ss += __shfl_xor(ss, 4, 32);
        ss += __shfl_xor(ss, 8, 32);
        if (n == 0) ssh[w][g * 8 + r] = ss;
    }
    __syncthreads();
    #pragma unroll
    for (int r = 0; r < 8; ++r) {
        const int m = g * 8 + r;
        float tot = ssh[0][m] + ssh[1][m] + ssh[2][m] + ssh[3][m];
        float inv = 1.0f / fmaxf(sqrtf(tot), 1e-12f);
        #pragma unroll
        for (int et = 0; et < 4; ++et) c[et][r] *= inv;
    }
}

// ---------------------------------------------------------------------------
// Elementwise helpers
// ---------------------------------------------------------------------------
__global__ void k_f32_to_f16(const float* __restrict__ src, _Float16* __restrict__ dst, int count) {
    int i = blockIdx.x * 256 + threadIdx.x;
    if (i < count) dst[i] = (_Float16)src[i];
}
__global__ void k_zero(float* __restrict__ p, int count) {
    int i = blockIdx.x * 256 + threadIdx.x;
    if (i < count) p[i] = 0.0f;
}
__global__ void k_scale(float* __restrict__ p, float s, int count) {
    int i = blockIdx.x * 256 + threadIdx.x;
    if (i < count) p[i] *= s;
}

// ---------------------------------------------------------------------------
// Phase A: accumulate mean of l2-normalized embedded rows per (inst,b)
// jobs: 0..2 = Ys(u) -> ybar_s ; 3..11 = Yp(u,v) -> ybar_p ; 12..20 = Xp -> xbar_p
// ---------------------------------------------------------------------------
__global__ void k_mean(const _Float16* __restrict__ U16,
                       const _Float16* __restrict__ Wsy16,
                       const _Float16* __restrict__ Wpy16,
                       const _Float16* __restrict__ Wpx16,
                       const float* __restrict__ bsy,
                       const float* __restrict__ bpy,
                       const float* __restrict__ bpx,
                       float* __restrict__ ybar_s,
                       float* __restrict__ ybar_p,
                       float* __restrict__ xbar_p) {
    __shared__ float ssh[4][16];
    const int job  = blockIdx.y;
    const int row0 = blockIdx.x * 16;
    const int b    = row0 >> 10;
    const int w    = threadIdx.x >> 5;

    const _Float16 *A, *W; const float* bias; float* outv;
    if (job < 3) {
        int u = job;
        A = U16 + (size_t)u * TOK * DD;
        W = Wsy16 + (size_t)u * DD * DD;
        bias = bsy + u * DD;
        outv = ybar_s + ((size_t)u * BB + b) * DD;
    } else if (job < 12) {
        int p = job - 3, v = p % 3;
        A = U16 + (size_t)v * TOK * DD;
        W = Wpy16 + (size_t)p * DD * DD;
        bias = bpy + p * DD;
        outv = ybar_p + ((size_t)p * BB + b) * DD;
    } else {
        int p = job - 12, u = p / 3;
        A = U16 + (size_t)u * TOK * DD;
        W = Wpx16 + (size_t)p * DD * DD;
        bias = bpx + p * DD;
        outv = xbar_p + ((size_t)p * BB + b) * DD;
    }

    v8f c[4];
    gemm4(A, W, bias, row0, w, c);
    l2norm_block(c, w, ssh);

    const int l = threadIdx.x & 31, n = l & 15, g = l >> 4;
    #pragma unroll
    for (int et = 0; et < 4; ++et) {
        float s = 0.0f;
        #pragma unroll
        for (int r = 0; r < 8; ++r) s += c[et][r];   // this lane's 8 tokens
        s += __shfl_xor(s, 16, 32);                  // combine the two groups
        if (g == 0) atomicAdd(&outv[w * 64 + et * 16 + n], s);
    }
}

// ---------------------------------------------------------------------------
// Phase B1: unary[u,b,n] = relu(U@Wu^T + bu) . wr + br
// ---------------------------------------------------------------------------
__global__ void k_unary(const _Float16* __restrict__ U16,
                        const _Float16* __restrict__ Wu16,
                        const float* __restrict__ bu,
                        const float* __restrict__ wr,
                        const float* __restrict__ br,
                        float* __restrict__ unary_pot) {
    __shared__ float dsh[4][16];
    const int u    = blockIdx.y;
    const int row0 = blockIdx.x * 16;
    const int w    = threadIdx.x >> 5;

    v8f c[4];
    gemm4(U16 + (size_t)u * TOK * DD, Wu16 + (size_t)u * DD * DD, bu + u * DD, row0, w, c);

    const int l = threadIdx.x & 31, n = l & 15, g = l >> 4;
    float wv[4];
    #pragma unroll
    for (int et = 0; et < 4; ++et) wv[et] = wr[u * DD + w * 64 + et * 16 + n];

    #pragma unroll
    for (int r = 0; r < 8; ++r) {
        float d = 0.0f;
        #pragma unroll
        for (int et = 0; et < 4; ++et) d += fmaxf(c[et][r], 0.0f) * wv[et];
        d += __shfl_xor(d, 1, 32);
        d += __shfl_xor(d, 2, 32);
        d += __shfl_xor(d, 4, 32);
        d += __shfl_xor(d, 8, 32);
        if (n == 0) dsh[w][g * 8 + r] = d;
    }
    __syncthreads();
    if (threadIdx.x < 16) {
        float tot = dsh[0][threadIdx.x] + dsh[1][threadIdx.x]
                  + dsh[2][threadIdx.x] + dsh[3][threadIdx.x];
        unary_pot[(size_t)u * TOK + row0 + threadIdx.x] = tot + br[u];
    }
}

// ---------------------------------------------------------------------------
// Phase B2: per-token dot of l2-normalized rows with a mean vector.
// jobs: 0..2 self (Xs . ybar_s) ; 3..11 pot_x (Xp . ybar_p) ; 12..20 pot_y (Yp . xbar_p)
// mean vectors are pre-scaled by 1/N so the dot is the row-mean of affinities.
// ---------------------------------------------------------------------------
__global__ void k_dot(const _Float16* __restrict__ U16,
                      const _Float16* __restrict__ Wsx16,
                      const _Float16* __restrict__ Wpx16,
                      const _Float16* __restrict__ Wpy16,
                      const float* __restrict__ bsx,
                      const float* __restrict__ bpx,
                      const float* __restrict__ bpy,
                      const float* __restrict__ ybar_s,
                      const float* __restrict__ ybar_p,
                      const float* __restrict__ xbar_p,
                      float* __restrict__ self_pot,
                      float* __restrict__ pot_x,
                      float* __restrict__ pot_y) {
    __shared__ float ssh[4][16];
    __shared__ float dsh[4][16];
    const int job  = blockIdx.y;
    const int row0 = blockIdx.x * 16;
    const int b    = row0 >> 10;
    const int w    = threadIdx.x >> 5;

    const _Float16 *A, *W; const float *bias, *vec; float* pot;
    if (job < 3) {
        int u = job;
        A = U16 + (size_t)u * TOK * DD;
        W = Wsx16 + (size_t)u * DD * DD;
        bias = bsx + u * DD;
        vec = ybar_s + ((size_t)u * BB + b) * DD;
        pot = self_pot + (size_t)u * TOK;
    } else if (job < 12) {
        int p = job - 3, u = p / 3;
        A = U16 + (size_t)u * TOK * DD;
        W = Wpx16 + (size_t)p * DD * DD;
        bias = bpx + p * DD;
        vec = ybar_p + ((size_t)p * BB + b) * DD;
        pot = pot_x + (size_t)p * TOK;
    } else {
        int p = job - 12, v = p % 3;
        A = U16 + (size_t)v * TOK * DD;
        W = Wpy16 + (size_t)p * DD * DD;
        bias = bpy + p * DD;
        vec = xbar_p + ((size_t)p * BB + b) * DD;
        pot = pot_y + (size_t)p * TOK;
    }

    v8f c[4];
    gemm4(A, W, bias, row0, w, c);
    l2norm_block(c, w, ssh);

    const int l = threadIdx.x & 31, n = l & 15, g = l >> 4;
    float vv[4];
    #pragma unroll
    for (int et = 0; et < 4; ++et) vv[et] = vec[w * 64 + et * 16 + n];

    #pragma unroll
    for (int r = 0; r < 8; ++r) {
        float d = 0.0f;
        #pragma unroll
        for (int et = 0; et < 4; ++et) d += c[et][r] * vv[et];
        d += __shfl_xor(d, 1, 32);
        d += __shfl_xor(d, 2, 32);
        d += __shfl_xor(d, 4, 32);
        d += __shfl_xor(d, 8, 32);
        if (n == 0) dsh[w][g * 8 + r] = d;
    }
    __syncthreads();
    if (threadIdx.x < 16) {
        pot[row0 + threadIdx.x] = dsh[0][threadIdx.x] + dsh[1][threadIdx.x]
                                + dsh[2][threadIdx.x] + dsh[3][threadIdx.x];
    }
}

// ---------------------------------------------------------------------------
// Phase C: combine potentials, softmax over tokens, attend over fp32 U.
// one block per (i, j, b); 256 threads (thread d = channel for attend).
// ---------------------------------------------------------------------------
__global__ void k_reduce(const float* __restrict__ u0, const float* __restrict__ u1,
                         const float* __restrict__ u2,
                         const float* __restrict__ unary_pot,
                         const float* __restrict__ self_pot,
                         const float* __restrict__ pot_x,
                         const float* __restrict__ pot_y,
                         const float* __restrict__ wdiag,
                         const float* __restrict__ wpair,
                         float* __restrict__ out) {
    __shared__ float red[NN];
    __shared__ float sm[8];

    const int idx = blockIdx.x;
    const int b = idx & 15;
    const int ij = idx >> 4;
    const int i = ij / 3, j = ij % 3;
    const float* Uj = (j == 0) ? u0 : (j == 1) ? u1 : u2;
    const int tid = threadIdx.x;
    const size_t base = (size_t)b * NN;

    if (i == j) {
        const int k1 = (i == 0) ? 1 : 0;
        const int k2 = (i == 2) ? 1 : 2;
        const float w0 = wdiag[i * 4 + 0], w1 = wdiag[i * 4 + 1];
        const float w2 = wdiag[i * 4 + 2], w3 = wdiag[i * 4 + 3];
        for (int n = tid; n < NN; n += 256) {
            size_t t = base + n;
            red[n] = w0 * unary_pot[(size_t)i * TOK + t] + w1 * self_pot[(size_t)i * TOK + t]
                   + w2 * pot_x[(size_t)(i * 3 + k1) * TOK + t]
                   + w3 * pot_x[(size_t)(i * 3 + k2) * TOK + t];
        }
    } else {
        const float* wp = wpair + (i * 3 + j) * 3;
        const float w0 = wp[0], w1 = wp[1], w2 = wp[2];
        for (int n = tid; n < NN; n += 256) {
            size_t t = base + n;
            red[n] = w0 * unary_pot[(size_t)j * TOK + t] + w1 * self_pot[(size_t)j * TOK + t]
                   + w2 * pot_y[(size_t)(i * 3 + j) * TOK + t];
        }
    }
    __syncthreads();

    // softmax max
    float m = -3.4e38f;
    for (int n = tid; n < NN; n += 256) m = fmaxf(m, red[n]);
    for (int s = 1; s < 32; s <<= 1) m = fmaxf(m, __shfl_xor(m, s, 32));
    if ((tid & 31) == 0) sm[tid >> 5] = m;
    __syncthreads();
    float M = sm[0];
    #pragma unroll
    for (int q = 1; q < 8; ++q) M = fmaxf(M, sm[q]);
    __syncthreads();

    // exp + sum
    float s = 0.0f;
    for (int n = tid; n < NN; n += 256) { float e = __expf(red[n] - M); red[n] = e; s += e; }
    for (int sh = 1; sh < 32; sh <<= 1) s += __shfl_xor(s, sh, 32);
    if ((tid & 31) == 0) sm[tid >> 5] = s;
    __syncthreads();
    float S = 0.0f;
    #pragma unroll
    for (int q = 0; q < 8; ++q) S += sm[q];
    const float inv = 1.0f / S;

    // attend: out[i,j,b,d] = (1/S) * sum_n e[n] * U[j][b,n,d]
    const int d = tid;
    float acc = 0.0f;
    const float* Ub = Uj + base * DD;
    for (int n = 0; n < NN; ++n) acc += red[n] * Ub[(size_t)n * DD + d];
    out[((size_t)ij * BB + b) * DD + d] = acc * inv;
}

// ---------------------------------------------------------------------------
extern "C" void kernel_launch(void* const* d_in, const int* in_sizes, int n_in,
                              void* d_out, int out_size, void* d_ws, size_t ws_size,
                              hipStream_t stream) {
    (void)in_sizes; (void)n_in; (void)out_size; (void)ws_size;

    const float* u0    = (const float*)d_in[0];
    const float* u1    = (const float*)d_in[1];
    const float* u2    = (const float*)d_in[2];
    const float* Wu    = (const float*)d_in[3];
    const float* bu    = (const float*)d_in[4];
    const float* wr    = (const float*)d_in[5];
    const float* br    = (const float*)d_in[6];
    const float* Wsx   = (const float*)d_in[7];
    const float* bsx   = (const float*)d_in[8];
    const float* Wsy   = (const float*)d_in[9];
    const float* bsy   = (const float*)d_in[10];
    const float* Wpx   = (const float*)d_in[11];
    const float* bpx   = (const float*)d_in[12];
    const float* Wpy   = (const float*)d_in[13];
    const float* bpy   = (const float*)d_in[14];
    const float* wdiag = (const float*)d_in[15];
    const float* wpair = (const float*)d_in[16];
    float* out = (float*)d_out;

    // ---- workspace carve-up (256-B aligned) ----
    char* ws = (char*)d_ws;
    size_t off = 0;
    auto alloc = [&](size_t bytes) -> char* {
        char* r = ws + off;
        off += (bytes + 255) & ~(size_t)255;
        return r;
    };
    _Float16* U16   = (_Float16*)alloc((size_t)3 * TOK * DD * sizeof(_Float16));
    _Float16* Wu16  = (_Float16*)alloc((size_t)3 * DD * DD * sizeof(_Float16));
    _Float16* Wsx16 = (_Float16*)alloc((size_t)3 * DD * DD * sizeof(_Float16));
    _Float16* Wsy16 = (_Float16*)alloc((size_t)3 * DD * DD * sizeof(_Float16));
    _Float16* Wpx16 = (_Float16*)alloc((size_t)9 * DD * DD * sizeof(_Float16));
    _Float16* Wpy16 = (_Float16*)alloc((size_t)9 * DD * DD * sizeof(_Float16));
    const int MEANS = 3 * BB * DD + 9 * BB * DD + 9 * BB * DD;   // 86016 floats
    float* means  = (float*)alloc((size_t)MEANS * sizeof(float));
    float* ybar_s = means;
    float* ybar_p = means + 3 * BB * DD;
    float* xbar_p = means + 3 * BB * DD + 9 * BB * DD;
    float* unary_pot = (float*)alloc((size_t)3 * TOK * sizeof(float));
    float* self_pot  = (float*)alloc((size_t)3 * TOK * sizeof(float));
    float* pot_x     = (float*)alloc((size_t)9 * TOK * sizeof(float));
    float* pot_y     = (float*)alloc((size_t)9 * TOK * sizeof(float));

    // ---- 1. fp32 -> f16 conversions ----
    const int CU = TOK * DD;            // 4,194,304 per util
    const int CW3 = 3 * DD * DD;        // 196,608
    const int CW9 = 9 * DD * DD;        // 589,824
    k_f32_to_f16<<<(CU + 255) / 256, 256, 0, stream>>>(u0, U16 + (size_t)0 * CU, CU);
    k_f32_to_f16<<<(CU + 255) / 256, 256, 0, stream>>>(u1, U16 + (size_t)1 * CU, CU);
    k_f32_to_f16<<<(CU + 255) / 256, 256, 0, stream>>>(u2, U16 + (size_t)2 * CU, CU);
    k_f32_to_f16<<<(CW3 + 255) / 256, 256, 0, stream>>>(Wu,  Wu16,  CW3);
    k_f32_to_f16<<<(CW3 + 255) / 256, 256, 0, stream>>>(Wsx, Wsx16, CW3);
    k_f32_to_f16<<<(CW3 + 255) / 256, 256, 0, stream>>>(Wsy, Wsy16, CW3);
    k_f32_to_f16<<<(CW9 + 255) / 256, 256, 0, stream>>>(Wpx, Wpx16, CW9);
    k_f32_to_f16<<<(CW9 + 255) / 256, 256, 0, stream>>>(Wpy, Wpy16, CW9);

    // ---- 2. zero mean accumulators ----
    k_zero<<<(MEANS + 255) / 256, 256, 0, stream>>>(means, MEANS);

    // ---- 3. phase A: means of normalized embedded rows ----
    k_mean<<<dim3(TILES, 21), 128, 0, stream>>>(
        U16, Wsy16, Wpy16, Wpx16, bsy, bpy, bpx, ybar_s, ybar_p, xbar_p);

    // ---- 4. scale to means (1/N) ----
    k_scale<<<(MEANS + 255) / 256, 256, 0, stream>>>(means, 1.0f / (float)NN, MEANS);

    // ---- 5. phase B: per-token potentials ----
    k_unary<<<dim3(TILES, 3), 128, 0, stream>>>(
        U16, Wu16, bu, wr, br, unary_pot);
    k_dot<<<dim3(TILES, 21), 128, 0, stream>>>(
        U16, Wsx16, Wpx16, Wpy16, bsx, bpx, bpy,
        ybar_s, ybar_p, xbar_p, self_pot, pot_x, pot_y);

    // ---- 6. phase C: combine + softmax + attend ----
    k_reduce<<<NUU * NUU * BB, 256, 0, stream>>>(
        u0, u1, u2, unary_pot, self_pot, pot_x, pot_y, wdiag, wpair, out);
}
